// BiMambaTranslator_79559974191142
// MI455X (gfx1250) — matrix operand order repaired and checked by reading, not compiled, not run on hardware
//
#include <hip/hip_runtime.h>
#include <hip/hip_bf16.h>
#include <cstdint>
#include <cstddef>

// ---------------------------------------------------------------- types ----
typedef _Float16 v16h  __attribute__((ext_vector_type(16)));
typedef _Float16 half8 __attribute__((ext_vector_type(8)));
typedef float    v8f   __attribute__((ext_vector_type(8)));

// ------------------------------------------------------------- constants ---
namespace cfg {
constexpr int V   = 16000;
constexpr int DM  = 512;
constexpr int DS  = 16;
constexpr int DC  = 4;
constexpr int DI  = 1024;   // EXP * DM
constexpr int DTR = 32;     // DM / 16
constexpr int DFF = 2048;
constexpr int NH  = 8;
constexpr int Bb  = 8;
constexpr int T   = 256;    // TS == TT
constexpr int M   = Bb * T; // 2048 tokens
}

__device__ __forceinline__ float siluf(float x) { return x / (1.f + __expf(-x)); }
__device__ __forceinline__ float geluf(float x) { return 0.5f * x * (1.f + erff(x * 0.70710678118654752f)); }

// ---------------------------------------------------------------- kernels --
//
// C[M,N] = A[M,K](f16) * Bt[K,N](f16) + bias.
// Block = 4 waves; each wave owns a 32(M) x 64(N) tile (2 A frags x 4 B frags,
// 8 v_wmma per K-step). The B tile (32 K-rows x 64 cols) is shared by all four
// waves, so it is staged into LDS with the CDNA5 async-to-LDS engine
// (global_load_async_to_lds_b128, ASYNCcnt, s_wait_asynccnt), double buffered.
// Requires: M % 128 == 0, N % 64 == 0, K % 32 == 0 (true for every call site).
__global__ void __launch_bounds__(128) gemm_wmma_k(
    const _Float16* __restrict__ A, const _Float16* __restrict__ Bt,
    const float* __restrict__ bias, float* __restrict__ C,
    int M, int N, int K, int hasBias)
{
  __shared__ _Float16 bsm[2][32 * 72];   // 32 K-rows x 64 halves, row stride 72 (bank pad)

  const int wave = threadIdx.x >> 5;
  const int lane = threadIdx.x & 31;
  const int l16  = lane & 15;
  const int hi   = lane >> 4;
  const int n0   = blockIdx.x * 64;
  const int m0   = (blockIdx.y * 4 + wave) * 32;

  const _Float16* a0 = A + (size_t)(m0 + l16) * K + hi * 8;
  const _Float16* a1 = a0 + (size_t)16 * K;

  // async-stage one 32xN64 slice of B into LDS buffer `buf`
  auto stage = [&](int kb, int buf) {
    int ci = threadIdx.x;                 // 256 16-byte chunks, 2 per thread
#pragma unroll
    for (int rep = 0; rep < 2; ++rep, ci += 128) {
      int row = ci >> 3, col = ci & 7;    // 8 chunks per 128-byte row
      unsigned lds = (unsigned)(uintptr_t)&bsm[buf][row * 72 + col * 8];
      const _Float16* g = Bt + (size_t)(kb + row) * N + n0 + col * 8;
      asm volatile("global_load_async_to_lds_b128 %0, %1, off"
                   :: "v"(lds), "v"(g) : "memory");
    }
  };

  v8f acc[2][4] = {};
  stage(0, 0);
  for (int kb = 0; kb < K; kb += 32) {
    const int cur = (kb >> 5) & 1;
    if (kb + 32 < K) {
      stage(kb + 32, cur ^ 1);                         // prefetch next tile
      asm volatile("s_wait_asynccnt 0x2" ::: "memory"); // current tile landed
    } else {
      asm volatile("s_wait_asynccnt 0x0" ::: "memory");
    }
    __syncthreads();

    union { v16h v; half8 h[2]; } ua0, ua1, ub[4];
    ua0.h[0] = *(const half8*)(a0 + kb);
    ua0.h[1] = *(const half8*)(a0 + kb + 16);
    ua1.h[0] = *(const half8*)(a1 + kb);
    ua1.h[1] = *(const half8*)(a1 + kb + 16);
    __builtin_prefetch(a0 + kb + 64, 0, 1);            // global_prefetch_b8
    __builtin_prefetch(a1 + kb + 64, 0, 1);
    const _Float16* brow = &bsm[cur][l16];             // lane = N-col, elements = K-rows (table order)
#pragma unroll
    for (int c = 0; c < 4; ++c) {
#pragma unroll
      for (int s = 0; s < 8; ++s) {
        ub[c].h[0][s] = brow[(8 * hi + s) * 72 + c * 16];
        ub[c].h[1][s] = brow[(16 + 8 * hi + s) * 72 + c * 16];
      }
    }
#pragma unroll
    for (int c = 0; c < 4; ++c) {
      acc[0][c] = __builtin_amdgcn_wmma_f32_16x16x32_f16(
          false, ua0.v, false, ub[c].v, (short)0, acc[0][c], false, false);
      acc[1][c] = __builtin_amdgcn_wmma_f32_16x16x32_f16(
          false, ua1.v, false, ub[c].v, (short)0, acc[1][c], false, false);
    }
    __syncthreads();
  }

#pragma unroll
  for (int smt = 0; smt < 2; ++smt) {
#pragma unroll
    for (int c = 0; c < 4; ++c) {
      const int n = n0 + c * 16 + l16;
      const float bv = hasBias ? bias[n] : 0.f;
#pragma unroll
      for (int r = 0; r < 8; ++r)
        C[(size_t)(m0 + smt * 16 + r + 8 * hi) * N + n] = acc[smt][c][r] + bv;
    }
  }
}

__global__ void cvt_f16_k(const float* __restrict__ s, _Float16* __restrict__ d, size_t n) {
  size_t i = (size_t)blockIdx.x * blockDim.x + threadIdx.x;
  if (i < n) d[i] = (_Float16)s[i];
}

// take cols [off, off+cols) of an [M, stride] f32 matrix -> compact [M, cols] f16
__global__ void cvt_slice_f16_k(const float* __restrict__ s, _Float16* __restrict__ d,
                                int M, int stride, int off, int cols) {
  size_t i = (size_t)blockIdx.x * blockDim.x + threadIdx.x;
  if (i >= (size_t)M * cols) return;
  int r = (int)(i / cols), c = (int)(i % cols);
  d[i] = (_Float16)s[(size_t)r * stride + off + c];
}

// W[N,K] f32 -> Wt[K,N] f16 (so GEMM computes x @ W.T)
__global__ void wtrans_f16_k(const float* __restrict__ W, _Float16* __restrict__ Wt,
                             int N, int K) {
  size_t i = (size_t)blockIdx.x * blockDim.x + threadIdx.x;
  if (i >= (size_t)N * K) return;
  int n = (int)(i / K), k = (int)(i % K);
  Wt[(size_t)k * N + n] = (_Float16)W[i];
}

__global__ void embed_k(const int* __restrict__ tok, const float* __restrict__ emb,
                        float* __restrict__ x, int ntok) {
  size_t i = (size_t)blockIdx.x * blockDim.x + threadIdx.x;
  if (i >= (size_t)ntok * cfg::DM) return;
  int r = (int)(i / cfg::DM), c = (int)(i % cfg::DM);
  x[i] = emb[(size_t)tok[r] * cfg::DM + c];
}

__global__ void __launch_bounds__(128) layernorm_k(
    const float* __restrict__ x, const float* __restrict__ g,
    const float* __restrict__ b, float* __restrict__ o, int ntok) {
  __shared__ float red[128];
  int t = blockIdx.x;
  if (t >= ntok) return;
  const float* xr = x + (size_t)t * cfg::DM;
  float s = 0.f;
  for (int i = threadIdx.x; i < cfg::DM; i += 128) s += xr[i];
  red[threadIdx.x] = s; __syncthreads();
  for (int st = 64; st > 0; st >>= 1) {
    if ((int)threadIdx.x < st) red[threadIdx.x] += red[threadIdx.x + st];
    __syncthreads();
  }
  float mean = red[0] / (float)cfg::DM;
  __syncthreads();
  float v = 0.f;
  for (int i = threadIdx.x; i < cfg::DM; i += 128) { float d = xr[i] - mean; v += d * d; }
  red[threadIdx.x] = v; __syncthreads();
  for (int st = 64; st > 0; st >>= 1) {
    if ((int)threadIdx.x < st) red[threadIdx.x] += red[threadIdx.x + st];
    __syncthreads();
  }
  float rstd = rsqrtf(red[0] / (float)cfg::DM + 1e-5f);
  float* orow = o + (size_t)t * cfg::DM;
  for (int i = threadIdx.x; i < cfg::DM; i += 128)
    orow[i] = (xr[i] - mean) * rstd * g[i] + b[i];
}

// depthwise causal conv (DC=4) over xi = first DI cols of xiz [M, 2*DI], then SiLU
__global__ void conv_silu_k(const float* __restrict__ xiz, const float* __restrict__ cw,
                            const float* __restrict__ cb, float* __restrict__ xs,
                            int B_, int T_) {
  size_t i = (size_t)blockIdx.x * blockDim.x + threadIdx.x;
  size_t n = (size_t)B_ * T_ * cfg::DI;
  if (i >= n) return;
  int d = (int)(i % cfg::DI);
  int t = (int)((i / cfg::DI) % T_);
  int b = (int)(i / ((size_t)cfg::DI * T_));
  float acc = cb[d];
#pragma unroll
  for (int j = 0; j < cfg::DC; ++j) {
    int tt = t - (cfg::DC - 1) + j;
    if (tt >= 0)
      acc += cw[d * cfg::DC + j] * xiz[(size_t)(b * T_ + tt) * (2 * cfg::DI) + d];
  }
  xs[i] = siluf(acc);
}

__global__ void softplus_k(float* __restrict__ x, size_t n) {
  size_t i = (size_t)blockIdx.x * blockDim.x + threadIdx.x;
  if (i >= n) return;
  float v = x[i];
  x[i] = (v > 20.f) ? v : log1pf(__expf(v));
}

// selective scan: one thread per (b,d) channel, DS states in registers.
__global__ void mamba_scan_k(const float* __restrict__ dtb, const float* __restrict__ xs,
                             const float* __restrict__ dbc, const float* __restrict__ A_log,
                             const float* __restrict__ Dp, float* __restrict__ ys,
                             int B_, int T_) {
  int id = blockIdx.x * blockDim.x + threadIdx.x;
  if (id >= B_ * cfg::DI) return;
  int b = id / cfg::DI, d = id % cfg::DI;
  float An[cfg::DS], h[cfg::DS];
#pragma unroll
  for (int s = 0; s < cfg::DS; ++s) { An[s] = -__expf(A_log[d * cfg::DS + s]); h[s] = 0.f; }
  float Dv = Dp[d];
  for (int t = 0; t < T_; ++t) {
    size_t row = (size_t)b * T_ + t;
    float dtv = dtb[row * cfg::DI + d];
    float xv  = xs[row * cfg::DI + d];
    const float* bc = dbc + row * 64;   // [dt(32) | B(16) | C(16)]
    float y = 0.f;
#pragma unroll
    for (int s = 0; s < cfg::DS; ++s) {
      h[s] = __expf(dtv * An[s]) * h[s] + dtv * bc[32 + s] * xv;
      y   += h[s] * bc[48 + s];
    }
    ys[row * cfg::DI + d] = y + Dv * xv;
  }
}

// g = ys * silu(z) where z = cols [DI, 2*DI) of xiz
__global__ void gate_mul_k(const float* __restrict__ ys, const float* __restrict__ xiz,
                           float* __restrict__ g, int M_) {
  size_t i = (size_t)blockIdx.x * blockDim.x + threadIdx.x;
  if (i >= (size_t)M_ * cfg::DI) return;
  int r = (int)(i / cfg::DI), c = (int)(i % cfg::DI);
  float z = xiz[(size_t)r * (2 * cfg::DI) + cfg::DI + c];
  g[i] = ys[i] * siluf(z);
}

__global__ void add_k(float* __restrict__ x, const float* __restrict__ y, size_t n) {
  size_t i = (size_t)blockIdx.x * blockDim.x + threadIdx.x;
  if (i < n) x[i] += y[i];
}

__global__ void gelu_k(float* __restrict__ x, size_t n) {
  size_t i = (size_t)blockIdx.x * blockDim.x + threadIdx.x;
  if (i < n) x[i] = geluf(x[i]);
}

__global__ void flip_copy_k(const float* __restrict__ s, float* __restrict__ d,
                            int B_, int T_, int D_) {
  size_t i = (size_t)blockIdx.x * blockDim.x + threadIdx.x;
  if (i >= (size_t)B_ * T_ * D_) return;
  int dd = (int)(i % D_);
  int t  = (int)((i / D_) % T_);
  int b  = (int)(i / ((size_t)T_ * D_));
  d[i] = s[((size_t)b * T_ + (T_ - 1 - t)) * D_ + dd];
}

__global__ void add_flip_k(float* __restrict__ x, const float* __restrict__ y,
                           int B_, int T_, int D_) {
  size_t i = (size_t)blockIdx.x * blockDim.x + threadIdx.x;
  if (i >= (size_t)B_ * T_ * D_) return;
  int dd = (int)(i % D_);
  int t  = (int)((i / D_) % T_);
  int b  = (int)(i / ((size_t)T_ * D_));
  x[i] += y[((size_t)b * T_ + (T_ - 1 - t)) * D_ + dd];
}

// online-softmax cross attention; one block = one (b, head, query) row, 64 threads = dh.
__global__ void __launch_bounds__(64) cross_attn_k(
    const float* __restrict__ q, const float* __restrict__ k,
    const float* __restrict__ v, float* __restrict__ o, int B_, int Tq, int Tk) {
  __shared__ float red[64];
  int blk = blockIdx.x;
  int qi  = blk % Tq;
  int hh  = (blk / Tq) % cfg::NH;
  int b   = blk / (Tq * cfg::NH);
  int t   = threadIdx.x;                 // head-dim index 0..63
  const int dh = cfg::DM / cfg::NH;      // 64
  float qv = q[((size_t)(b * Tq + qi)) * cfg::DM + hh * dh + t] * 0.125f; // 1/sqrt(64)
  float m = -1e30f, l = 0.f, acc = 0.f;
  for (int kk = 0; kk < Tk; ++kk) {
    size_t kr = ((size_t)(b * Tk + kk)) * cfg::DM + hh * dh + t;
    red[t] = qv * k[kr];
    __syncthreads();
    for (int s2 = 32; s2 > 0; s2 >>= 1) {
      if (t < s2) red[t] += red[t + s2];
      __syncthreads();
    }
    float sc = red[0];
    __syncthreads();
    float mn  = fmaxf(m, sc);
    float esc = __expf(m - mn);
    float pe  = __expf(sc - mn);
    l   = l * esc + pe;
    acc = acc * esc + pe * v[kr];
    m = mn;
  }
  o[((size_t)(b * Tq + qi)) * cfg::DM + hh * dh + t] = acc / l;
}

// ------------------------------------------------------------- host side ---
namespace {

using namespace cfg;

inline unsigned gdx(size_t n, unsigned b) { return (unsigned)((n + b - 1) / b); }

struct Scratch {
  _Float16 *a16, *w16;
  float *xiz, *xs, *dbc, *dtb, *ys, *gate;
  float *h, *h2, *mo, *xenc, *xdec, *encout, *q, *k, *v, *hidden;
};

Scratch carve(void* ws) {
  char* p = (char*)ws;
  auto take = [&](size_t bytes) -> void* {
    void* r = (void*)p;
    p += (bytes + 255) & ~(size_t)255;
    return r;
  };
  Scratch s;
  s.a16    = (_Float16*)take((size_t)M * DFF * sizeof(_Float16)); // largest f16 activation
  s.w16    = (_Float16*)take((size_t)V * DM * sizeof(_Float16));  // largest weight (emb^T)
  s.xiz    = (float*)take((size_t)M * 2 * DI * 4);
  s.xs     = (float*)take((size_t)M * DI * 4);
  s.dbc    = (float*)take((size_t)M * 64 * 4);
  s.dtb    = (float*)take((size_t)M * DI * 4);
  s.ys     = (float*)take((size_t)M * DI * 4);
  s.gate   = (float*)take((size_t)M * DI * 4);
  s.h      = (float*)take((size_t)M * DM * 4);
  s.h2     = (float*)take((size_t)M * DM * 4);
  s.mo     = (float*)take((size_t)M * DM * 4);
  s.xenc   = (float*)take((size_t)M * DM * 4);
  s.xdec   = (float*)take((size_t)M * DM * 4);
  s.encout = (float*)take((size_t)M * DM * 4);
  s.q      = (float*)take((size_t)M * DM * 4);
  s.k      = (float*)take((size_t)M * DM * 4);
  s.v      = (float*)take((size_t)M * DM * 4);
  s.hidden = (float*)take((size_t)M * DFF * 4);
  return s;
}

void launch_gemm(hipStream_t st, const _Float16* A, const _Float16* Bt,
                 const float* bias, float* C, int m, int n, int k) {
  dim3 grid(n / 64, m / 128);   // wave tile 32x64, 4 waves/block
  gemm_wmma_k<<<grid, 128, 0, st>>>(A, Bt, bias, C, m, n, k, bias ? 1 : 0);
}

struct MambaW {
  const float *in_proj, *conv_w, *conv_b, *x_proj, *dt_w, *dt_b, *A_log, *D, *out_proj;
};

MambaW mamba_at(void* const* d_in, int base) {
  MambaW w;
  w.in_proj  = (const float*)d_in[base + 0];
  w.conv_w   = (const float*)d_in[base + 1];
  w.conv_b   = (const float*)d_in[base + 2];
  w.x_proj   = (const float*)d_in[base + 3];
  w.dt_w     = (const float*)d_in[base + 4];
  w.dt_b     = (const float*)d_in[base + 5];
  w.A_log    = (const float*)d_in[base + 6];
  w.D        = (const float*)d_in[base + 7];
  w.out_proj = (const float*)d_in[base + 8];
  return w;
}

void run_mamba(hipStream_t st, const float* hin, float* out, const MambaW& w, const Scratch& s) {
  // in_proj: [M,512] -> [M,2048]
  cvt_f16_k<<<gdx((size_t)M * DM, 256), 256, 0, st>>>(hin, s.a16, (size_t)M * DM);
  wtrans_f16_k<<<gdx((size_t)2 * DI * DM, 256), 256, 0, st>>>(w.in_proj, s.w16, 2 * DI, DM);
  launch_gemm(st, s.a16, s.w16, nullptr, s.xiz, M, 2 * DI, DM);
  // causal depthwise conv + SiLU
  conv_silu_k<<<gdx((size_t)M * DI, 256), 256, 0, st>>>(s.xiz, w.conv_w, w.conv_b, s.xs, Bb, T);
  // x_proj: [M,1024] -> [M,64]
  cvt_f16_k<<<gdx((size_t)M * DI, 256), 256, 0, st>>>(s.xs, s.a16, (size_t)M * DI);
  wtrans_f16_k<<<gdx((size_t)64 * DI, 256), 256, 0, st>>>(w.x_proj, s.w16, 64, DI);
  launch_gemm(st, s.a16, s.w16, nullptr, s.dbc, M, 64, DI);
  // dt = softplus(dt_raw @ dt_w.T + dt_b): [M,32] -> [M,1024]
  cvt_slice_f16_k<<<gdx((size_t)M * DTR, 256), 256, 0, st>>>(s.dbc, s.a16, M, 64, 0, DTR);
  wtrans_f16_k<<<gdx((size_t)DI * DTR, 256), 256, 0, st>>>(w.dt_w, s.w16, DI, DTR);
  launch_gemm(st, s.a16, s.w16, w.dt_b, s.dtb, M, DI, DTR);
  softplus_k<<<gdx((size_t)M * DI, 256), 256, 0, st>>>(s.dtb, (size_t)M * DI);
  // selective scan
  mamba_scan_k<<<gdx((size_t)Bb * DI, 256), 256, 0, st>>>(s.dtb, s.xs, s.dbc, w.A_log, w.D, s.ys, Bb, T);
  // y * silu(z), out_proj
  gate_mul_k<<<gdx((size_t)M * DI, 256), 256, 0, st>>>(s.ys, s.xiz, s.gate, M);
  cvt_f16_k<<<gdx((size_t)M * DI, 256), 256, 0, st>>>(s.gate, s.a16, (size_t)M * DI);
  wtrans_f16_k<<<gdx((size_t)DM * DI, 256), 256, 0, st>>>(w.out_proj, s.w16, DM, DI);
  launch_gemm(st, s.a16, s.w16, nullptr, out, M, DM, DI);
}

void run_ffn(hipStream_t st, const float* hin, float* x_acc,
             const float* w1, const float* b1, const float* w2, const float* b2,
             const Scratch& s) {
  cvt_f16_k<<<gdx((size_t)M * DM, 256), 256, 0, st>>>(hin, s.a16, (size_t)M * DM);
  wtrans_f16_k<<<gdx((size_t)DFF * DM, 256), 256, 0, st>>>(w1, s.w16, DFF, DM);
  launch_gemm(st, s.a16, s.w16, b1, s.hidden, M, DFF, DM);
  gelu_k<<<gdx((size_t)M * DFF, 256), 256, 0, st>>>(s.hidden, (size_t)M * DFF);
  cvt_f16_k<<<gdx((size_t)M * DFF, 256), 256, 0, st>>>(s.hidden, s.a16, (size_t)M * DFF);
  wtrans_f16_k<<<gdx((size_t)DM * DFF, 256), 256, 0, st>>>(w2, s.w16, DM, DFF);
  launch_gemm(st, s.a16, s.w16, b2, s.mo, M, DM, DFF);
  add_k<<<gdx((size_t)M * DM, 256), 256, 0, st>>>(x_acc, s.mo, (size_t)M * DM);
}

} // namespace

extern "C" void kernel_launch(void* const* d_in, const int* in_sizes, int n_in,
                              void* d_out, int out_size, void* d_ws, size_t ws_size,
                              hipStream_t stream) {
  (void)in_sizes; (void)n_in; (void)out_size; (void)ws_size;
  using namespace cfg;
  const int* src = (const int*)d_in[0];
  const int* tgt = (const int*)d_in[1];
  const float* emb = (const float*)d_in[3];
#define F(i) ((const float*)d_in[(i)])
  Scratch s = carve(d_ws);
  hipStream_t st = stream;

  // ----------------- encoder -----------------
  embed_k<<<gdx((size_t)M * DM, 256), 256, 0, st>>>(src, emb, s.xenc, M);
  for (int lyr = 0; lyr < 6; ++lyr) {
    int base = 4 + 26 * lyr;
    // h = LN1(x)  (pad mask all-false -> keep == 1)
    layernorm_k<<<M, 128, 0, st>>>(s.xenc, F(base + 0), F(base + 1), s.h, M);
    // forward mamba
    run_mamba(st, s.h, s.mo, mamba_at(d_in, base + 2), s);
    add_k<<<gdx((size_t)M * DM, 256), 256, 0, st>>>(s.xenc, s.mo, (size_t)M * DM);
    // backward mamba on time-flipped h
    flip_copy_k<<<gdx((size_t)M * DM, 256), 256, 0, st>>>(s.h, s.h2, Bb, T, DM);
    run_mamba(st, s.h2, s.mo, mamba_at(d_in, base + 11), s);
    add_flip_k<<<gdx((size_t)M * DM, 256), 256, 0, st>>>(s.xenc, s.mo, Bb, T, DM);
    // FFN
    layernorm_k<<<M, 128, 0, st>>>(s.xenc, F(base + 20), F(base + 21), s.h, M);
    run_ffn(st, s.h, s.xenc, F(base + 22), F(base + 23), F(base + 24), F(base + 25), s);
  }
  layernorm_k<<<M, 128, 0, st>>>(s.xenc, F(160), F(161), s.encout, M);

  // ----------------- decoder -----------------
  embed_k<<<gdx((size_t)M * DM, 256), 256, 0, st>>>(tgt, emb, s.xdec, M);
  for (int lyr = 0; lyr < 6; ++lyr) {
    int base = 162 + 27 * lyr;
    // self mamba
    layernorm_k<<<M, 128, 0, st>>>(s.xdec, F(base + 0), F(base + 1), s.h, M);
    run_mamba(st, s.h, s.mo, mamba_at(d_in, base + 2), s);
    add_k<<<gdx((size_t)M * DM, 256), 256, 0, st>>>(s.xdec, s.mo, (size_t)M * DM);
    // cross attention
    layernorm_k<<<M, 128, 0, st>>>(s.xdec, F(base + 11), F(base + 12), s.h, M);
    cvt_f16_k<<<gdx((size_t)M * DM, 256), 256, 0, st>>>(s.encout, s.a16, (size_t)M * DM);
    wtrans_f16_k<<<gdx((size_t)DM * DM, 256), 256, 0, st>>>(F(base + 15), s.w16, DM, DM); // wk
    launch_gemm(st, s.a16, s.w16, F(base + 16), s.k, M, DM, DM);
    wtrans_f16_k<<<gdx((size_t)DM * DM, 256), 256, 0, st>>>(F(base + 17), s.w16, DM, DM); // wv
    launch_gemm(st, s.a16, s.w16, F(base + 18), s.v, M, DM, DM);
    cvt_f16_k<<<gdx((size_t)M * DM, 256), 256, 0, st>>>(s.h, s.a16, (size_t)M * DM);
    wtrans_f16_k<<<gdx((size_t)DM * DM, 256), 256, 0, st>>>(F(base + 13), s.w16, DM, DM); // wq
    launch_gemm(st, s.a16, s.w16, F(base + 14), s.q, M, DM, DM);
    cross_attn_k<<<Bb * NH * T, 64, 0, st>>>(s.q, s.k, s.v, s.gate, Bb, T, T);
    cvt_f16_k<<<gdx((size_t)M * DM, 256), 256, 0, st>>>(s.gate, s.a16, (size_t)M * DM);
    wtrans_f16_k<<<gdx((size_t)DM * DM, 256), 256, 0, st>>>(F(base + 19), s.w16, DM, DM); // wo
    launch_gemm(st, s.a16, s.w16, F(base + 20), s.mo, M, DM, DM);
    add_k<<<gdx((size_t)M * DM, 256), 256, 0, st>>>(s.xdec, s.mo, (size_t)M * DM);
    // FFN
    layernorm_k<<<M, 128, 0, st>>>(s.xdec, F(base + 21), F(base + 22), s.h, M);
    run_ffn(st, s.h, s.xdec, F(base + 23), F(base + 24), F(base + 25), F(base + 26), s);
  }

  // final norm + logits = x @ emb.T  (M=2048, N=16000, K=512)
  layernorm_k<<<M, 128, 0, st>>>(s.xdec, F(324), F(325), s.h, M);
  cvt_f16_k<<<gdx((size_t)M * DM, 256), 256, 0, st>>>(s.h, s.a16, (size_t)M * DM);
  wtrans_f16_k<<<gdx((size_t)V * DM, 256), 256, 0, st>>>(emb, s.w16, V, DM);
  launch_gemm(st, s.a16, s.w16, nullptr, (float*)d_out, M, V, DM);
#undef F
}
